// CachedModel_22153441313361
// MI455X (gfx1250) — compile-verified, matrix operand order checked by reading, not run on hardware
//
#include <hip/hip_runtime.h>

typedef __attribute__((ext_vector_type(16))) __bf16 v16bf;
typedef __attribute__((ext_vector_type(8)))  float  v8f;

#define TCN_H   512
#define TCN_B   128
#define STRIDE  516
#define EPSF    1e-8f

__device__ __forceinline__ v8f wmma_bf16(v16bf a, v16bf b, v8f c) {
  // D = A(16x32 bf16) * B(32x16 bf16) + C(16x16 f32)
  return __builtin_amdgcn_wmma_f32_16x16x32_bf16(false, a, false, b,
                                                 (short)0, c, false, false);
}

// ---------------------------------------------------------------------------
// Pack a row-major f32 matrix [M][K] (M padded to Mpad, zero-filled) into
// bf16 A-fragment layout: Wf[((ks*MT + mt)*32 + lane)*16 + j], so each lane
// reads its whole v16bf fragment as 32 contiguous bytes.
// Lane map (16-bit A 16x32): m = lane&15, klocal = (lane>>4)*8 + (j&7) + (j&8)*2
// ---------------------------------------------------------------------------
__global__ __launch_bounds__(256) void pack_a_kernel(
    const float* __restrict__ W, __bf16* __restrict__ Wf,
    int M, int K, int Mpad)
{
  const int idx = blockIdx.x * 256 + threadIdx.x;
  if (idx >= Mpad * K) return;
  const int j    = idx & 15;
  const int lane = (idx >> 4) & 31;
  const int rest = idx >> 9;            // ks*MT + mt
  const int MT   = Mpad >> 4;
  const int ks   = rest / MT;
  const int mt   = rest - ks * MT;
  const int mloc = lane & 15;
  const int hb   = lane >> 4;
  const int kloc = (hb << 3) + (j & 7) + ((j & 8) << 1);
  const int m = mt * 16 + mloc;
  const int k = ks * 32 + kloc;
  const float v = (m < M) ? W[(size_t)m * K + k] : 0.0f;
  Wf[idx] = (__bf16)v;
}

// ---------------------------------------------------------------------------
// Encoder: Conv1d(2->128, k=50, s=50) + ReLU  -> enc[128][8]
// ---------------------------------------------------------------------------
__global__ __launch_bounds__(256) void encoder_kernel(
    const float* __restrict__ mix, const float* __restrict__ Wenc,
    float* __restrict__ enc)
{
  const int i = blockIdx.x * 256 + threadIdx.x;
  if (i >= 1024) return;
  const int n = i >> 3, j = i & 7;
  const float* w  = Wenc + n * 100;
  const float* m0 = mix + j * 50;
  const float* m1 = mix + 400 + j * 50;
  float s = 0.0f;
  for (int l = 0; l < 50; ++l) s += w[l] * m0[l] + w[50 + l] * m1[l];
  enc[n * 8 + j] = fmaxf(s, 0.0f);
}

// ---------------------------------------------------------------------------
// Ring shift: d_out buffers stay [C][516]; the TCN working buffer x0 is
// TRANSPOSED [T=516][C=128] so GEMM B-fragments / residuals are contiguous.
// ---------------------------------------------------------------------------
__global__ __launch_bounds__(256) void ring_shift_kernel(
    const float* __restrict__ ebuf_in, const float* __restrict__ enc,
    const float* __restrict__ lbuf_in, float* __restrict__ out_ebuf,
    float* __restrict__ out_lbuf, float* __restrict__ x0T)
{
  const int i = blockIdx.x * 256 + threadIdx.x;
  if (i >= TCN_B * STRIDE) return;
  const int c = i / STRIDE, t = i % STRIDE;
  out_ebuf[i] = (t < 508) ? ebuf_in[c * STRIDE + t + 8] : enc[c * 8 + (t - 508)];
  if (t < 508) {
    const float lv = lbuf_in[c * STRIDE + t + 8];
    out_lbuf[i] = lv;
    x0T[t * TCN_B + c] = lv;
  }
}

// ---------------------------------------------------------------------------
// Front-end: cLN(enc) + bottleneck (W_bn 128x128, packed bf16) via WMMA.
// ---------------------------------------------------------------------------
__global__ __launch_bounds__(256) void frontend_kernel(
    const float* __restrict__ enc, const float* __restrict__ lng,
    const float* __restrict__ lnb, const __bf16* __restrict__ wbnf,
    float* __restrict__ x0T, float* __restrict__ out_lbuf)
{
  __shared__ __align__(32) __bf16 sBf[4][32][16];
  __shared__ float sMu[8], sRs[8];
  const int tid  = threadIdx.x;
  const int wave = tid >> 5, lane = tid & 31;
  const int nidx = lane & 15, halfsel = lane >> 4;

  if (tid < 8) {
    float s = 0.0f, ss = 0.0f;
    for (int c = 0; c < 128; ++c) { const float v = enc[c * 8 + tid]; s += v; ss += v * v; }
    const float mu = s * (1.0f / 128.0f);
    sMu[tid] = mu;
    sRs[tid] = rsqrtf(ss * (1.0f / 128.0f) - mu * mu + EPSF);
  }
  __syncthreads();
  if (tid < 128) {                       // one thread per (ks,lane) B row
    const int ks = tid >> 5, ln = tid & 31;
    const int f = ln & 15, hb = ln >> 4;
    v16bf bv;
#pragma unroll
    for (int j = 0; j < 16; ++j) {
      const int c = ks * 32 + (hb << 3) + (j & 7) + ((j & 8) << 1);
      const float v = (f < 8) ? (lng[c] * (enc[c * 8 + f] - sMu[f]) * sRs[f] + lnb[c]) : 0.0f;
      bv[j] = (__bf16)v;
    }
    *(v16bf*)&sBf[ks][ln][0] = bv;
  }
  __syncthreads();

  v8f acc = {};
#pragma unroll
  for (int ks = 0; ks < 4; ++ks) {
    const v16bf a = *(const v16bf*)&wbnf[((ks * 8 + wave) * 32 + lane) * 16];
    const v16bf b = *(const v16bf*)&sBf[ks][lane][0];
    acc = wmma_bf16(a, b, acc);
  }
  if (nidx < 8) {
#pragma unroll
    for (int r = 0; r < 8; ++r) {
      const int m = wave * 16 + (halfsel << 3) + r;
      const float v = acc[r];
      x0T[(508 + nidx) * TCN_B + m] = v;
      out_lbuf[m * STRIDE + 508 + nidx] = v;
    }
  }
}

// ---------------------------------------------------------------------------
// One TCN block, x in TRANSPOSED layout [T][128].
// One workgroup (8 waves) per 16-frame output tile; Z kept in registers.
// ---------------------------------------------------------------------------
__global__ __launch_bounds__(256) void tcn_block_kernel(
    const float* __restrict__ xin, float* __restrict__ xout,
    const __bf16* __restrict__ w1f, int mtbase1,
    const float* __restrict__ dwc,
    const float* __restrict__ p1p, const float* __restrict__ g1,
    const float* __restrict__ b1,  const float* __restrict__ p2p,
    const float* __restrict__ g2,  const float* __restrict__ b2,
    const __bf16* __restrict__ w2f, int mtbase2,
    int d, int T_in, int T_out)
{
  __shared__ __align__(32) __bf16 sY2f[16][32][16];  // W2 B-fragments (16KB)
  __shared__ float sSum[16], sSqs[16], sMu[16], sRs[16];

  const int tid     = threadIdx.x;
  const int wave    = tid >> 5;
  const int lane    = tid & 31;
  const int nidx    = lane & 15;      // frame column within tile
  const int halfsel = lane >> 4;
  const int t0      = blockIdx.x * 16;
  const float p1v = p1p[0], p2v = p2p[0];

  // Prefetch this block's packed W2 fragments (read-once; hide the fetch
  // behind the 3-tap W1 phase instead of staging read-once data in LDS).
  {
    const char* w2base = (const char*)&w2f[(size_t)mtbase2 * 32 * 16];
    __builtin_prefetch(w2base + (size_t)tid * 512, 0, 1);
    __builtin_prefetch(w2base + (size_t)tid * 512 + 256, 0, 1);
    __builtin_prefetch(g2 + tid * 2, 0, 1);
    __builtin_prefetch(b2 + tid * 2, 0, 1);
  }

  float zacc[4][8];                   // dwconv accumulator, registers only
#pragma unroll
  for (int mt = 0; mt < 4; ++mt)
#pragma unroll
    for (int r = 0; r < 8; ++r) zacc[mt][r] = 0.0f;

  for (int tap = 0; tap < 3; ++tap) {
    if (tid < 16) { sSum[tid] = 0.0f; sSqs[tid] = 0.0f; }
    __syncthreads();

    // B fragments: x column (c0+nidx), channels contiguous in [T][128] layout
    const int col = t0 + tap * d + nidx;
    v16bf bfr[4];
#pragma unroll
    for (int ks = 0; ks < 4; ++ks)
#pragma unroll
      for (int j = 0; j < 16; ++j) bfr[ks][j] = (__bf16)0.0f;
    if (col < T_in) {
      const float* xp = xin + (size_t)col * TCN_B + (halfsel << 3);
#pragma unroll
      for (int ks = 0; ks < 4; ++ks) {
        const float4 q0 = *(const float4*)(xp + ks * 32);
        const float4 q1 = *(const float4*)(xp + ks * 32 + 4);
        const float4 q2 = *(const float4*)(xp + ks * 32 + 16);
        const float4 q3 = *(const float4*)(xp + ks * 32 + 20);
        bfr[ks][0]=(__bf16)q0.x;  bfr[ks][1]=(__bf16)q0.y;
        bfr[ks][2]=(__bf16)q0.z;  bfr[ks][3]=(__bf16)q0.w;
        bfr[ks][4]=(__bf16)q1.x;  bfr[ks][5]=(__bf16)q1.y;
        bfr[ks][6]=(__bf16)q1.z;  bfr[ks][7]=(__bf16)q1.w;
        bfr[ks][8]=(__bf16)q2.x;  bfr[ks][9]=(__bf16)q2.y;
        bfr[ks][10]=(__bf16)q2.z; bfr[ks][11]=(__bf16)q2.w;
        bfr[ks][12]=(__bf16)q3.x; bfr[ks][13]=(__bf16)q3.y;
        bfr[ks][14]=(__bf16)q3.z; bfr[ks][15]=(__bf16)q3.w;
      }
    }

    // W1 GEMM: each wave covers 4 tiles of 16 rows of H=512
    v8f accs[4];
#pragma unroll
    for (int mt = 0; mt < 4; ++mt) {
      v8f acc = {};
#pragma unroll
      for (int ks = 0; ks < 4; ++ks) {
        const v16bf a = *(const v16bf*)
            &w1f[(((size_t)ks * 448 + mtbase1 + wave * 4 + mt) * 32 + lane) * 16];
        acc = wmma_bf16(a, bfr[ks], acc);
      }
      float s = 0.0f, ss = 0.0f;
#pragma unroll
      for (int r = 0; r < 8; ++r) {
        float v = acc[r];
        v = (v >= 0.0f) ? v : p1v * v;        // PReLU in registers
        acc[r] = v;
        s += v; ss += v * v;
      }
      atomicAdd(&sSum[nidx], s);
      atomicAdd(&sSqs[nidx], ss);
      accs[mt] = acc;
    }
    __syncthreads();
    if (tid < 16) {
      const float mu = sSum[tid] * (1.0f / 512.0f);
      sMu[tid] = mu;
      sRs[tid] = rsqrtf(sSqs[tid] * (1.0f / 512.0f) - mu * mu + EPSF);
    }
    __syncthreads();
    // cLN + depthwise tap weight, accumulate into register Z
    const float mu = sMu[nidx];
    const float rs = sRs[nidx];
#pragma unroll
    for (int mt = 0; mt < 4; ++mt) {
      const int h0 = (wave * 4 + mt) * 16 + (halfsel << 3);
#pragma unroll
      for (int r = 0; r < 8; ++r) {
        const int h = h0 + r;
        const float yn = g1[h] * (accs[mt][r] - mu) * rs + b1[h];
        zacc[mt][r] += dwc[h * 3 + tap] * yn;
      }
    }
  }

  // --- PReLU(p2) + cLN(g2,b2), write directly into B-fragment layout ------
  if (tid < 16) { sSum[tid] = 0.0f; sSqs[tid] = 0.0f; }
  __syncthreads();
  {
    float s = 0.0f, ss = 0.0f;
#pragma unroll
    for (int mt = 0; mt < 4; ++mt)
#pragma unroll
      for (int r = 0; r < 8; ++r) {
        float v = zacc[mt][r];
        v = (v >= 0.0f) ? v : p2v * v;
        zacc[mt][r] = v;
        s += v; ss += v * v;
      }
    atomicAdd(&sSum[nidx], s);
    atomicAdd(&sSqs[nidx], ss);
  }
  __syncthreads();
  if (tid < 16) {
    const float mu = sSum[tid] * (1.0f / 512.0f);
    sMu[tid] = mu;
    sRs[tid] = rsqrtf(sSqs[tid] * (1.0f / 512.0f) - mu * mu + EPSF);
  }
  __syncthreads();
  {
    const float mu = sMu[nidx];
    const float rs = sRs[nidx];
#pragma unroll
    for (int mt = 0; mt < 4; ++mt) {
      const int h0 = (wave * 4 + mt) * 16 + (halfsel << 3);
#pragma unroll
      for (int r = 0; r < 8; ++r) {
        const int h = h0 + r;
        const float v = g2[h] * (zacc[mt][r] - mu) * rs + b2[h];
        const int ks = h >> 5, kloc = h & 31;
        const int hb = (kloc & 15) >> 3;
        const int jb = (kloc & 7) + ((kloc >> 4) << 3);
        sY2f[ks][nidx + (hb << 4)][jb] = (__bf16)v;
      }
    }
  }
  __syncthreads();

  // --- out = W2 @ Y2 + residual (transposed layout: contiguous per lane) --
  {
    v8f acc = {};
#pragma unroll 4
    for (int ks = 0; ks < 16; ++ks) {
      const v16bf a = *(const v16bf*)
          &w2f[(((size_t)ks * 112 + mtbase2 + wave) * 32 + lane) * 16];
      const v16bf b = *(const v16bf*)&sY2f[ks][lane][0];
      acc = wmma_bf16(a, b, acc);
    }
    const int t = t0 + nidx;
    if (t < T_out) {
      const int c0 = wave * 16 + (halfsel << 3);
      const float* pi = xin + (size_t)(t + 2 * d) * TCN_B + c0;
      float* po = xout + (size_t)t * TCN_B + c0;
#pragma unroll
      for (int r = 0; r < 8; ++r) po[r] = pi[r] + acc[r];
    }
  }
}

// ---------------------------------------------------------------------------
// Mask conv + ReLU, mask newest 8 encoder frames, decoder (50x128 padded 64).
// ---------------------------------------------------------------------------
__global__ __launch_bounds__(256) void maskdec_kernel(
    const float* __restrict__ xfT, const __bf16* __restrict__ wmaskf,
    const float* __restrict__ enc, const __bf16* __restrict__ wdecf,
    float* __restrict__ est)
{
  __shared__ __align__(32) __bf16 sXf[4][32][16];
  __shared__ __align__(32) __bf16 sSf[4][32][16];
  const int tid  = threadIdx.x;
  const int wave = tid >> 5, lane = tid & 31;
  const int nidx = lane & 15, halfsel = lane >> 4;

  if (tid < 128) {
    const int ks = tid >> 5, ln = tid & 31;
    const int f = ln & 15, hb = ln >> 4;
    v16bf bv;
#pragma unroll
    for (int j = 0; j < 16; ++j) {
      const int c = ks * 32 + (hb << 3) + (j & 7) + ((j & 8) << 1);
      bv[j] = (__bf16)((f < 8) ? xfT[f * TCN_B + c] : 0.0f);
    }
    *(v16bf*)&sXf[ks][ln][0] = bv;
  }
  __syncthreads();

  // mask = ReLU(W_mask @ x)
  v8f acc = {};
#pragma unroll
  for (int ks = 0; ks < 4; ++ks) {
    const v16bf a = *(const v16bf*)&wmaskf[((ks * 8 + wave) * 32 + lane) * 16];
    const v16bf b = *(const v16bf*)&sXf[ks][lane][0];
    acc = wmma_bf16(a, b, acc);
  }
  // masked encoder frames -> decoder B-fragment layout
#pragma unroll
  for (int r = 0; r < 8; ++r) {
    const int m = wave * 16 + (halfsel << 3) + r;   // encoder channel
    const float mk = fmaxf(acc[r], 0.0f);
    const float v  = (nidx < 8) ? mk * enc[m * 8 + nidx] : 0.0f;
    const int ks = m >> 5, kloc = m & 31;
    const int hb = (kloc & 15) >> 3;
    const int jb = (kloc & 7) + ((kloc >> 4) << 3);
    sSf[ks][nidx + (hb << 4)][jb] = (__bf16)v;
  }
  __syncthreads();

  if (wave < 4) {                       // uniform per-wave branch; EXEC full
    v8f acc2 = {};
#pragma unroll
    for (int ks = 0; ks < 4; ++ks) {
      const v16bf a = *(const v16bf*)&wdecf[((ks * 4 + wave) * 32 + lane) * 16];
      const v16bf b = *(const v16bf*)&sSf[ks][lane][0];
      acc2 = wmma_bf16(a, b, acc2);
    }
    if (nidx < 8) {
#pragma unroll
      for (int r = 0; r < 8; ++r) {
        const int l = wave * 16 + (halfsel << 3) + r;
        if (l < 50) est[nidx * 50 + l] = acc2[r];
      }
    }
  }
}

// ---------------------------------------------------------------------------
extern "C" void kernel_launch(void* const* d_in, const int* in_sizes, int n_in,
                              void* d_out, int out_size, void* d_ws, size_t ws_size,
                              hipStream_t stream) {
  const float* mixture = (const float*)d_in[0];
  const float* ebuf_in = (const float*)d_in[1];
  const float* lbuf_in = (const float*)d_in[2];
  const float* W_enc   = (const float*)d_in[3];
  const float* ln_g    = (const float*)d_in[4];
  const float* ln_b    = (const float*)d_in[5];
  const float* W_bn    = (const float*)d_in[6];
  const float* blk_w1  = (const float*)d_in[7];
  const float* blk_p1  = (const float*)d_in[8];
  const float* blk_g1  = (const float*)d_in[9];
  const float* blk_b1  = (const float*)d_in[10];
  const float* blk_dw  = (const float*)d_in[11];
  const float* blk_p2  = (const float*)d_in[12];
  const float* blk_g2  = (const float*)d_in[13];
  const float* blk_b2  = (const float*)d_in[14];
  const float* blk_w2  = (const float*)d_in[15];
  const float* W_mask  = (const float*)d_in[16];
  const float* W_dec   = (const float*)d_in[17];

  float* est      = (float*)d_out;                 // [400]
  float* out_ebuf = est + 400;                     // [128*516]
  float* out_lbuf = out_ebuf + 128 * STRIDE;       // [128*516]

  float* ws     = (float*)d_ws;
  float* ws_enc = ws;                              // 1024
  float* x0     = ws + 1024;                       // 66048 (transposed ping)
  float* x1     = x0 + 128 * STRIDE;               // 66048 (transposed pong)

  __bf16* w1f    = (__bf16*)(x1 + 128 * STRIDE);   // 14*512*128
  __bf16* w2f    = w1f + 14 * 512 * 128;           // 14*128*512
  __bf16* wbnf   = w2f + 14 * 128 * 512;           // 128*128
  __bf16* wmaskf = wbnf + 128 * 128;               // 128*128
  __bf16* wdecf  = wmaskf + 128 * 128;             // 64*128

  // Per-call weight repack to bf16 fragment layout (deterministic, no cache).
  pack_a_kernel<<<(7168 * 128 + 255) / 256, 256, 0, stream>>>(blk_w1, w1f, 7168, 128, 7168);
  pack_a_kernel<<<(1792 * 512 + 255) / 256, 256, 0, stream>>>(blk_w2, w2f, 1792, 512, 1792);
  pack_a_kernel<<<(128 * 128 + 255) / 256, 256, 0, stream>>>(W_bn,   wbnf,   128, 128, 128);
  pack_a_kernel<<<(128 * 128 + 255) / 256, 256, 0, stream>>>(W_mask, wmaskf, 128, 128, 128);
  pack_a_kernel<<<(64 * 128 + 255) / 256, 256, 0, stream>>>(W_dec,  wdecf,   50, 128, 64);

  encoder_kernel<<<4, 256, 0, stream>>>(mixture, W_enc, ws_enc);
  ring_shift_kernel<<<(128 * STRIDE + 255) / 256, 256, 0, stream>>>(
      ebuf_in, ws_enc, lbuf_in, out_ebuf, out_lbuf, x0);
  frontend_kernel<<<1, 256, 0, stream>>>(ws_enc, ln_g, ln_b, wbnf, x0, out_lbuf);

  float* xin = x0;
  float* xout = x1;
  int T = 516;
  for (int i = 0; i < 14; ++i) {
    const int d = 1 << (i % 7);
    const int T_out = T - 2 * d;
    const int nblk = (T_out + 15) / 16;
    tcn_block_kernel<<<nblk, 256, 0, stream>>>(
        xin, xout,
        w1f, i * 32,
        blk_dw + (size_t)i * 512 * 3,
        blk_p1 + i, blk_g1 + (size_t)i * 512, blk_b1 + (size_t)i * 512,
        blk_p2 + i, blk_g2 + (size_t)i * 512, blk_b2 + (size_t)i * 512,
        w2f, i * 8,
        d, T, T_out);
    float* tmp = xin; xin = xout; xout = tmp;
    T = T_out;
  }

  maskdec_kernel<<<1, 256, 0, stream>>>(xin, wmaskf, ws_enc, wdecf, est);
}
